// ParaGraphLayer_58445914964479
// MI455X (gfx1250) — matrix operand order repaired and checked by reading, not compiled, not run on hardware
//
#include <hip/hip_runtime.h>

#define N_NODES 20000
#define DIM     512
#define N_EDGES 150000

typedef __attribute__((ext_vector_type(16))) __bf16 v16bf;
typedef __attribute__((ext_vector_type(2)))  __bf16 v2bf;
typedef __attribute__((ext_vector_type(8)))  float  v8f;

// pack two f32 -> two bf16 in one dword (hardware packed cvt when available)
__device__ inline unsigned pk2(float lo, float hi) {
#if __has_builtin(__builtin_amdgcn_cvt_pk_bf16_f32)
    v2bf v = __builtin_amdgcn_cvt_pk_bf16_f32(lo, hi);
    return __builtin_bit_cast(unsigned, v);
#else
    v2bf v;
    v[0] = (__bf16)lo;
    v[1] = (__bf16)hi;
    return __builtin_bit_cast(unsigned, v);
#endif
}

// ---------------------------------------------------------------------------
// Tiled bf16 WMMA GEMM:  C[M,NN] = op(A) @ B   (A f32, converted while staging)
// CONCAT: A = [x | acc+bias], K=1024; each 32-wide K-tile lies in one half.
// Block tile 128x64, 8 waves (4x2), wave tile 32x32 = 2x2 WMMA 16x16x32 frags.
// Double-buffered LDS; B staged TRANSPOSED so fragments read contiguously.
// All dims compile-time so address math strength-reduces to shifts/immediates.
// ---------------------------------------------------------------------------
template <bool CONCAT, bool RELU, int M, int K, int NN>
__global__ __launch_bounds__(256) void gemm_bf16_wmma(
    const float* __restrict__ A0, const float* __restrict__ A1,
    const float* __restrict__ bias, const float* __restrict__ B,
    float* __restrict__ C) {
    constexpr int LDA = 40;   // A tile 128 rows x 32 k (bf16), padded
    constexpr int LDB = 40;   // B tile stored [col][k]: 64 cols x 32 k, padded
    __shared__ __bf16 sA[2][128 * LDA];
    __shared__ __bf16 sB[2][64 * LDB];

    const int t       = threadIdx.x;
    const int rowBase = blockIdx.y * 128;
    const int colBase = blockIdx.x * 64;
    const int wid     = t >> 5, lane = t & 31;
    const int waveR   = wid & 3, waveC = wid >> 2;
    const int hl      = lane >> 4;        // lane half
    const int mm      = lane & 15;

    // staging assignments
    const int arow = t >> 1;              // 0..127
    const int acol = (t & 1) * 16;        // 0 or 16
    const int growA = rowBase + arow;
    const int br   = t >> 4;              // 0..15 -> k rows 2br, 2br+1
    const int bcg  = (t & 15) * 4;        // col group of 4

    float4 areg[4];                       // 16 A values
    float4 breg[2];                       // 2 k-rows x 4 cols

    auto loadRegs = [&](int kt) {
        const int k0 = kt * 32;
        const float* Ap = A0;
        int kk = k0;
        bool ub = false;
        if (CONCAT && k0 >= DIM) { Ap = A1; kk = k0 - DIM; ub = true; }
        if (growA < M) {
            const float4* p = (const float4*)(Ap + (long)growA * DIM + kk + acol);
#pragma unroll
            for (int j = 0; j < 4; ++j) areg[j] = p[j];
            if constexpr (CONCAT) {
                if (ub) {
                    const float4* bp = (const float4*)(bias + kk + acol);
#pragma unroll
                    for (int j = 0; j < 4; ++j) {
                        float4 bv = bp[j];
                        areg[j].x += bv.x; areg[j].y += bv.y;
                        areg[j].z += bv.z; areg[j].w += bv.w;
                    }
                }
            }
        } else {
#pragma unroll
            for (int j = 0; j < 4; ++j) areg[j] = make_float4(0.f, 0.f, 0.f, 0.f);
        }
        const float* bp0 = B + (long)(k0 + 2 * br) * NN + colBase + bcg;
        breg[0] = *(const float4*)bp0;
        breg[1] = *(const float4*)(bp0 + NN);
    };

    auto storeLDS = [&](int buf) {
        unsigned* pa = (unsigned*)&sA[buf][arow * LDA + acol];
#pragma unroll
        for (int j = 0; j < 4; ++j) {
            pa[2 * j + 0] = pk2(((const float*)&areg[j])[0], ((const float*)&areg[j])[1]);
            pa[2 * j + 1] = pk2(((const float*)&areg[j])[2], ((const float*)&areg[j])[3]);
        }
        const float* b0 = (const float*)&breg[0];
        const float* b1 = (const float*)&breg[1];
        unsigned* pb = (unsigned*)&sB[buf][0];
#pragma unroll
        for (int j = 0; j < 4; ++j) {
            const int col = bcg + j;
            pb[(col * LDB + 2 * br) >> 1] = pk2(b0[j], b1[j]);  // [col][k] pair
        }
    };

    v8f acc[2][2] = {};

    auto compute = [&](int buf) {
        v16bf af[2], bfg[2];
#pragma unroll
        for (int ti = 0; ti < 2; ++ti) {
            // A 16x32 layout: runs K = 8*hl..+7 and 16+8*hl..+7 (contiguous)
            const __bf16* p = &sA[buf][(waveR * 32 + ti * 16 + mm) * LDA + 8 * hl];
#pragma unroll
            for (int e = 0; e < 8; ++e) af[ti][e] = p[e];
#pragma unroll
            for (int e = 0; e < 8; ++e) af[ti][8 + e] = p[16 + e];
        }
#pragma unroll
        for (int tj = 0; tj < 2; ++tj) {
            // B 32x16 layout: lane = N col; runs K = 8*hl..+7 and 16+8*hl..+7 (transposed tile)
            const __bf16* p = &sB[buf][(waveC * 32 + tj * 16 + mm) * LDB + 8 * hl];
#pragma unroll
            for (int e = 0; e < 8; ++e) bfg[tj][e] = p[e];
#pragma unroll
            for (int e = 0; e < 8; ++e) bfg[tj][8 + e] = p[16 + e];
        }
#pragma unroll
        for (int ti = 0; ti < 2; ++ti)
#pragma unroll
            for (int tj = 0; tj < 2; ++tj)
                acc[ti][tj] = __builtin_amdgcn_wmma_f32_16x16x32_bf16(
                    false, af[ti], false, bfg[tj], (short)0, acc[ti][tj],
                    false, false);
    };

    constexpr int KT = K / 32;
    loadRegs(0);
    storeLDS(0);
#pragma unroll 2
    for (int kt = 0; kt < KT; ++kt) {
        __syncthreads();
        const int cur = kt & 1;
        if (kt + 1 < KT) loadRegs(kt + 1);   // global loads in flight over WMMA
        compute(cur);
        if (kt + 1 < KT) storeLDS(1 - cur);
    }

    // epilogue: C/D layout: vgpr r -> M = r + 8*half, lane%16 -> N
    const int row0 = rowBase + waveR * 32 + hl * 8;
    const int col0 = colBase + waveC * 32 + mm;
    float* cbase = C + (long)row0 * NN + col0;
    if (rowBase + 128 <= M) {
        // fast path: whole block in range, immediate-offset stores
#pragma unroll
        for (int ti = 0; ti < 2; ++ti)
#pragma unroll
            for (int tj = 0; tj < 2; ++tj)
#pragma unroll
                for (int r = 0; r < 8; ++r) {
                    float v = acc[ti][tj][r];
                    if (RELU) v = fmaxf(v, 0.f);
                    cbase[(long)(ti * 16 + r) * NN + tj * 16] = v;
                }
    } else {
#pragma unroll
        for (int ti = 0; ti < 2; ++ti)
#pragma unroll
            for (int tj = 0; tj < 2; ++tj)
#pragma unroll
                for (int r = 0; r < 8; ++r) {
                    if (row0 + ti * 16 + r < M) {
                        float v = acc[ti][tj][r];
                        if (RELU) v = fmaxf(v, 0.f);
                        cbase[(long)(ti * 16 + r) * NN + tj * 16] = v;
                    }
                }
    }
}

// ---------------------------------------------------------------------------
// Helpers: zero / per-relation softmax state init
// ---------------------------------------------------------------------------
__global__ void k_zero(float* __restrict__ p, long n) {
    long i = blockIdx.x * (long)blockDim.x + threadIdx.x;
    if (i < n) p[i] = 0.f;
}

__global__ void k_init_md(float* __restrict__ m, float* __restrict__ den, int n) {
    int i = blockIdx.x * blockDim.x + threadIdx.x;
    if (i < n) { m[i] = -INFINITY; den[i] = 0.f; }
}

// el = h . al ; er = h . ar   (one wave per node, float4 lanes, shfl reduce)
__global__ __launch_bounds__(256) void k_eler(
    const float* __restrict__ h, const float* __restrict__ al,
    const float* __restrict__ ar, float* __restrict__ el,
    float* __restrict__ er, int n) {
    const int node = blockIdx.x * 8 + (threadIdx.x >> 5);
    const int lane = threadIdx.x & 31;
    if (node >= n) return;
    const float4* hp  = (const float4*)(h + (long)node * DIM);
    const float4* alp = (const float4*)al;
    const float4* arp = (const float4*)ar;
    float sl = 0.f, sr = 0.f;
#pragma unroll
    for (int r = 0; r < 4; ++r) {
        const int idx = lane + r * 32;
        float4 hv = hp[idx], av = alp[idx], bv = arp[idx];
        sl += hv.x * av.x + hv.y * av.y + hv.z * av.z + hv.w * av.w;
        sr += hv.x * bv.x + hv.y * bv.y + hv.z * bv.z + hv.w * bv.w;
    }
#pragma unroll
    for (int off = 16; off > 0; off >>= 1) {
        sl += __shfl_xor(sl, off, 32);
        sr += __shfl_xor(sr, off, 32);
    }
    if (lane == 0) { el[node] = sl; er[node] = sr; }
}

// e = leaky_relu(el[src]+er[dst]); segment-max via ordered-int atomic trick
__global__ void k_edge_logits(const int* __restrict__ eg,
                              const float* __restrict__ el,
                              const float* __restrict__ er,
                              float* __restrict__ ew, float* __restrict__ m,
                              int ne) {
    int e = blockIdx.x * blockDim.x + threadIdx.x;
    if (e >= ne) return;
    const int s = eg[e], d = eg[ne + e];
    float v = el[s] + er[d];
    v = v > 0.f ? v : 0.2f * v;
    ew[e] = v;
    if (v >= 0.f)
        atomicMax((int*)m + d, __float_as_int(v));
    else
        atomicMin((unsigned int*)m + d, (unsigned int)__float_as_int(v));
}

__global__ void k_fix_m(float* __restrict__ m, int n) {
    int i = blockIdx.x * blockDim.x + threadIdx.x;
    if (i < n && !(m[i] > -1e37f)) m[i] = 0.f;  // -inf (empty segment) -> 0
}

// w = exp(e - m[dst]); denom[dst] += w
__global__ void k_edge_w(const int* __restrict__ eg, float* __restrict__ ew,
                         const float* __restrict__ m, float* __restrict__ den,
                         int ne) {
    int e = blockIdx.x * blockDim.x + threadIdx.x;
    if (e >= ne) return;
    const int d = eg[ne + e];
    float w = __expf(ew[e] - m[d]);
    ew[e] = w;
    atomicAdd(den + d, w);
}

// acc[dst] += h[src] * alpha   (one wave per edge; 16 floats per lane)
__global__ __launch_bounds__(256) void k_scatter(
    const int* __restrict__ eg, const float* __restrict__ ew,
    const float* __restrict__ den, const float* __restrict__ h,
    float* __restrict__ acc, int ne) {
    const int edge = (int)((blockIdx.x * (long)blockDim.x + threadIdx.x) >> 5);
    const int lane = threadIdx.x & 31;
    if (edge >= ne) return;
    const int s = eg[edge], d = eg[ne + edge];
    float dn = den[d];
    dn = (dn == 0.f) ? 1.f : dn;
    const float alpha = ew[edge] / dn;
    const float4* hp = (const float4*)(h + (long)s * DIM);
    float* op = acc + (long)d * DIM;
#pragma unroll
    for (int r = 0; r < 4; ++r) {
        const int idx = lane + r * 32;
        float4 hv = hp[idx];
        const int base = idx * 4;
        atomicAdd(op + base + 0, hv.x * alpha);
        atomicAdd(op + base + 1, hv.y * alpha);
        atomicAdd(op + base + 2, hv.z * alpha);
        atomicAdd(op + base + 3, hv.w * alpha);
    }
}

// ---------------------------------------------------------------------------
extern "C" void kernel_launch(void* const* d_in, const int* in_sizes, int n_in,
                              void* d_out, int out_size, void* d_ws, size_t ws_size,
                              hipStream_t stream) {
    (void)in_sizes; (void)n_in; (void)out_size; (void)ws_size;

    const float* x    = (const float*)d_in[0];
    const float* Wl   = (const float*)d_in[17];
    const float* bias = (const float*)d_in[18];

    // workspace carve-up (all 256B aligned)
    const long ND4 = (long)N_NODES * DIM * 4;   // 40,960,000 B
    const long N4  = 80128;                     // N_NODES*4 padded
    const long E4  = 600064;                    // N_EDGES*4 padded
    char* w = (char*)d_ws;
    float* h    = (float*)w;  w += ND4;
    float* accP = (float*)w;  w += ND4;
    float* accQ = (float*)w;  w += ND4;
    float* el   = (float*)w;  w += N4;
    float* er   = (float*)w;  w += N4;
    float* m    = (float*)w;  w += N4;
    float* den  = (float*)w;  w += N4;
    float* ew   = (float*)w;  w += E4;

    const dim3 blk(256);
    const dim3 gG(DIM / 64, (N_NODES + 127) / 128);   // GEMM grid (8,157)
    const int  gN    = (N_NODES + 255) / 256;
    const int  gE    = (N_EDGES + 255) / 256;
    const int  gEw   = (N_EDGES * 32 + 255) / 256;    // wave-per-edge scatter
    const int  gEler = (N_NODES + 7) / 8;

    // out accumulator starts at zero
    k_zero<<<(int)(((long)N_NODES * DIM + 255) / 256), blk, 0, stream>>>(
        accP, (long)N_NODES * DIM);

    for (int i = 0; i < 4; ++i) {
        const int*   edges = (const int*)d_in[1 + 4 * i];
        const float* Wi    = (const float*)d_in[2 + 4 * i];
        const float* ali   = (const float*)d_in[3 + 4 * i];
        const float* ari   = (const float*)d_in[4 + 4 * i];

        // h = x @ W_i   (bf16 WMMA)
        gemm_bf16_wmma<false, false, N_NODES, DIM, DIM><<<gG, blk, 0, stream>>>(
            x, nullptr, nullptr, Wi, h);

        // attention logits per node
        k_eler<<<gEler, blk, 0, stream>>>(h, ali, ari, el, er, N_NODES);

        // edge softmax
        k_init_md<<<gN, blk, 0, stream>>>(m, den, N_NODES);
        k_edge_logits<<<gE, blk, 0, stream>>>(edges, el, er, ew, m, N_EDGES);
        k_fix_m<<<gN, blk, 0, stream>>>(m, N_NODES);
        k_edge_w<<<gE, blk, 0, stream>>>(edges, ew, m, den, N_EDGES);

        // message scatter: accP += h[src] * alpha
        k_scatter<<<gEw, blk, 0, stream>>>(edges, ew, den, h, accP, N_EDGES);

        // out = relu(cat(x, accP + bias) @ Wl)  (fused concat + bias + relu)
        float* outp = (i == 3) ? (float*)d_out : accQ;
        gemm_bf16_wmma<true, true, N_NODES, 2 * DIM, DIM><<<gG, blk, 0, stream>>>(
            x, accP, bias, Wl, outp);

        if (i < 3) { float* tmp = accP; accP = accQ; accQ = tmp; }
    }
}
